// SoftPixelRadiusCNN_62904091018198
// MI455X (gfx1250) — matrix operand order repaired and checked by reading, not compiled, not run on hardware
//
#include <hip/hip_runtime.h>

// CDNA5 / gfx1250: wave32, WMMA 16x16x32 f16 -> f32
typedef __attribute__((ext_vector_type(16))) _Float16 v16h;
typedef __attribute__((ext_vector_type(8)))  _Float16 v8h;
typedef __attribute__((ext_vector_type(8)))  float    v8f;
typedef __attribute__((ext_vector_type(4)))  float    v4f;

#define SUBDIV 3
#define KNBR   32
#define NFEAT  64
#define FEPS   1e-6f
#define WAVES_PER_BLOCK 8
#define KPAD   40   // padded k-stride (80B) -> keeps 16B alignment, breaks bank conflicts

// wave32 XOR-butterfly step via ds_swizzle (group-of-32 mode):
// imm = {1'b0, xor_mask[4:0], or_mask=0, and_mask=0x1f}
// Mask must be an integer constant expression -> template parameter.
template <int XMASK>
__device__ __forceinline__ float xor_swap_add(float x) {
    int yi = __builtin_amdgcn_ds_swizzle(__builtin_bit_cast(int, x),
                                         (XMASK << 10) | 0x1f);
    return x + __builtin_bit_cast(float, yi);
}

__global__ __launch_bounds__(256)
void soft_pixel_radius_wmma(const float* __restrict__ features,
                            const float* __restrict__ distsq,
                            const int*   __restrict__ nbr_idx,
                            const float* __restrict__ length_scale,
                            float*       __restrict__ out,
                            int V)
{
    // transposed neighbor-feature tile: [feature][neighbor] in f16
    __shared__ _Float16 s_tileT[WAVES_PER_BLOCK][NFEAT][KPAD];
    // radius-bin weights: rows 0..2 = w_i[k], row 3 = zero (D rows >=3 unread)
    __shared__ _Float16 s_w[WAVES_PER_BLOCK][4][KNBR];

    const int lane = threadIdx.x & 31;
    const int wave = threadIdx.x >> 5;
    const int v    = blockIdx.x * WAVES_PER_BLOCK + wave;
    if (v >= V) return;                       // uniform per wave: EXEC stays all-ones

    const float ls     = length_scale[0];
    const float rls    = __builtin_amdgcn_rcpf(ls);       // v_rcp_f32, hoisted
    const float scaler = 10.0f * ls * (float)SUBDIV;

    // ---- phase 1: lane = neighbor k -------------------------------------
    const int   k   = lane;
    const int   row_idx = nbr_idx[(size_t)v * KNBR + k];
    const float dsq = distsq[(size_t)v * KNBR + k];
    const float d   = sqrtf(dsq + FEPS);

    float rden[SUBDIV];
#pragma unroll
    for (int i = 0; i < SUBDIV; ++i) {
        const float t = d - (float)i * rls;
        const float w = __expf(-scaler * t * t);          // v_exp_f32 (TRANS)
        // wave-wide XOR butterfly: den_i = sum_k w_i[k] (all lanes get total)
        float s = w;
        s = xor_swap_add<16>(s);
        s = xor_swap_add<8>(s);
        s = xor_swap_add<4>(s);
        s = xor_swap_add<2>(s);
        s = xor_swap_add<1>(s);
        rden[i] = __builtin_amdgcn_rcpf(s + FEPS);        // fast rcp: den >= eps > 0
        s_w[wave][i][k] = (_Float16)w;
    }
    s_w[wave][3][k] = (_Float16)0.0f;

    // gather this lane's neighbor row (64 f32 = 16x b128), transpose into LDS f16
    const v4f* __restrict__ frow = (const v4f*)(features + (size_t)row_idx * NFEAT);
#pragma unroll
    for (int j = 0; j < NFEAT / 4; ++j) {
        const v4f x = frow[j];
#pragma unroll
        for (int c = 0; c < 4; ++c)
            s_tileT[wave][j * 4 + c][k] = (_Float16)x[c];
    }

    // same-wave producer->consumer: DS completion is enough, no barrier
    asm volatile("s_wait_dscnt 0" ::: "memory");

    // ---- phase 2: build A (16x32 f16) -----------------------------------
    // ISA A layout (16-bit, 16x32): lanes 0-15 hold M=lane, K=0..7 (v0-3) and
    // K=16..23 (v4-7); lanes 16-31 hold K=8..15 and K=24..31.
    // Row clamp (row & 3): rows 4..15 of A replicate rows 0..3, so D row 8+i
    // is a bitwise copy of D row i -> lanes 16-31 mirror lanes 0-15.
    const int row  = lane & 15;
    const int kbA  = (lane < 16) ? 0 : 8;
    const _Float16* wr = &s_w[wave][row & 3][0];          // broadcast reads, no conflicts
    const v8h a_lo = *(const v8h*)(wr + kbA);             // K = kbA..kbA+7
    const v8h a_hi = *(const v8h*)(wr + kbA + 16);        // K = kbA+16..kbA+23
    v16h A;
#pragma unroll
    for (int t = 0; t < 8; ++t) { A[t] = a_lo[t]; A[t + 8] = a_hi[t]; }

    // ---- phase 3: 4 feature blocks of 16 columns ------------------------
    // ISA B layout (16-bit, 32x16): lane holds column N=lane%16; lanes 0-15
    // carry K=0..7 and K=16..23, lanes 16-31 carry K=8..15 and K=24..31.
    const int ncol = lane & 15;
    const int kbB  = (lane < 16) ? 0 : 8;
    float* __restrict__ obase = out + (size_t)v * (SUBDIV * NFEAT) + ncol;
#pragma unroll
    for (int nb = 0; nb < 4; ++nb) {
        const int col = nb * 16 + ncol;
        const _Float16* bp = &s_tileT[wave][col][kbB];    // K = kbB..kbB+7, kbB+16..kbB+23
        const v8h b_lo = *(const v8h*)(bp);
        const v8h b_hi = *(const v8h*)(bp + 16);
        v16h B;
#pragma unroll
        for (int t = 0; t < 8; ++t) { B[t] = b_lo[t]; B[t + 8] = b_hi[t]; }

        v8f acc = {};
        acc = __builtin_amdgcn_wmma_f32_16x16x32_f16(
                  /*neg_a=*/false, A, /*neg_b=*/false, B,
                  /*c_mod=*/(short)0, acc, /*reuse_a=*/false, /*reuse_b=*/false);

        // D VGPR i: lanes 0-15 = row i, lanes 16-31 = identical copy (row 8+i).
        // All 32 lanes store; lane pairs write the same value to the same
        // address -> deterministic, and no EXEC manipulation around WMMA.
#pragma unroll
        for (int i = 0; i < SUBDIV; ++i)
            obase[i * NFEAT + nb * 16] = acc[i] * rden[i];
    }
}

extern "C" void kernel_launch(void* const* d_in, const int* in_sizes, int n_in,
                              void* d_out, int out_size, void* d_ws, size_t ws_size,
                              hipStream_t stream) {
    const float* features = (const float*)d_in[0];   // (V, 64) f32
    const float* distsq   = (const float*)d_in[1];   // (V, 32) f32
    const int*   nbr      = (const int*)  d_in[2];   // (V, 32) i32
    const float* ls       = (const float*)d_in[3];   // scalar
    float* out = (float*)d_out;                      // (V, 192) f32

    const int V = in_sizes[1] / KNBR;
    const int blocks = (V + WAVES_PER_BLOCK - 1) / WAVES_PER_BLOCK;
    soft_pixel_radius_wmma<<<blocks, 256, 0, stream>>>(features, distsq, nbr, ls, out, V);
}
